// MultiheadAttention_1846835937655
// MI455X (gfx1250) — compile-verified, matrix operand order checked by reading, not run on hardware
//
#include <hip/hip_runtime.h>

#ifndef __has_builtin
#define __has_builtin(x) 0
#endif

#if __has_builtin(__builtin_amdgcn_tensor_load_to_lds) && \
    __has_builtin(__builtin_amdgcn_s_wait_tensorcnt)
#define HAVE_TDM 1
#else
#define HAVE_TDM 0
#endif

typedef __attribute__((ext_vector_type(16))) __bf16 v16bf;
typedef __attribute__((ext_vector_type(8)))  float  v8f;
typedef __attribute__((ext_vector_type(4)))  unsigned int v4u;
typedef __attribute__((ext_vector_type(4)))  int v4i;
typedef __attribute__((ext_vector_type(8)))  int v8i;

static __device__ __forceinline__ unsigned lds_off(const void* p) {
  // Generic pointers to LDS carry the byte offset in the low 32 bits
  // (aperture check truncates to the low 32 bits, ISA 10.2).
  return (unsigned)(size_t)p;
}

#if HAVE_TDM
// Issue a TDM 2-D tile load (bf16 elements) global -> LDS.
// D# layout per CDNA5 ISA ch.8: group0 = {flags, lds_addr, global_addr, type=2},
// group1 = {data_size/pad, tensor dims, tile dims, dim0 stride}; groups 2/3 zero.
static __device__ __forceinline__ void tdm_load_2d_bf16(
    unsigned lds_byte_off, const void* gptr,
    unsigned tile_x, unsigned tile_y,
    unsigned long long tensor_x, unsigned long long tensor_y,
    unsigned long long stride_x,
    unsigned pad_interval_code, unsigned pad_amount_code, bool pad_en)
{
  const unsigned long long ga = (unsigned long long)(size_t)gptr;
  v4u g0 = { 1u,                                   // count=1, user-mode descriptor
             lds_byte_off,                         // lds_addr
             (unsigned)(ga & 0xffffffffull),       // global_addr[31:0]
             (unsigned)((ga >> 32) & 0x01ffffffull) | (2u << 30) };  // addr[56:32], type=2
  const unsigned w0 = (1u << 16)                                  // data_size = 2 bytes
                    | (pad_en ? (1u << 20) : 0u)
                    | (pad_interval_code << 22) | (pad_amount_code << 25);
  v8i g1 = { (int)w0,
             (int)((tensor_x & 0xffffull) << 16),                               // dim0 lo16
             (int)(((tensor_x >> 16) & 0xffffull) | ((tensor_y & 0xffffull) << 16)),
             (int)(((tensor_y >> 16) & 0xffffull) |
                   ((unsigned long long)(tile_x & 0xffffu) << 16)),             // tile_dim0
             (int)(tile_y & 0xffffu),                                           // tile_dim1
             (int)(stride_x & 0xffffffffull),                                   // stride0 lo32
             (int)((stride_x >> 32) & 0xffffull),                               // stride0 hi16
             0 };
  v4i gz = { 0, 0, 0, 0 };
#if __clang_major__ >= 23
  v8i gz8 = { 0, 0, 0, 0, 0, 0, 0, 0 };
  __builtin_amdgcn_tensor_load_to_lds(g0, g1, gz, gz, gz8, 0);
#else
  __builtin_amdgcn_tensor_load_to_lds(g0, g1, gz, gz, 0);
#endif
}
#endif  // HAVE_TDM

// Load a 16x32 bf16 WMMA operand fragment from a row-major [row][K] tile.
// ISA 7.12.2 (16-bit A 16x32): lane l (0..15), g=lane/16: row M=l,
// K runs {8g..8g+7} and {16+8g..16+8g+7} -> two 16B loads.
static __device__ __forceinline__ v16bf frag_ld(const __bf16* t, int stride, int lane) {
  const int l = lane & 15;
  const int g = (lane >> 4) & 1;
  const __bf16* p = t + (long)l * stride + 8 * g;
  v16bf f;
  uint4* u = reinterpret_cast<uint4*>(&f);
  u[0] = *reinterpret_cast<const uint4*>(p);
  u[1] = *reinterpret_cast<const uint4*>(p + 16);
  return f;
}

// ---------------- conversion kernels ----------------
__global__ void mha_cvt_bf16(const float* __restrict__ in, __bf16* __restrict__ out, long n) {
  long i  = (long)blockIdx.x * blockDim.x + threadIdx.x;
  long st = (long)gridDim.x * blockDim.x;
  for (; i < n; i += st) out[i] = (__bf16)in[i];
}

// in: [K][N] f32 (row-major)  ->  out: [N][K] bf16 (row-major)
__global__ void mha_transpose_bf16(const float* __restrict__ in, __bf16* __restrict__ out,
                                   int K, int N) {
  long total = (long)K * N;
  long i  = (long)blockIdx.x * blockDim.x + threadIdx.x;
  long st = (long)gridDim.x * blockDim.x;
  for (; i < total; i += st) {
    long kk = i / N, nn = i % N;
    out[nn * (long)K + kk] = (__bf16)in[i];
  }
}

// ---------------- tiled bf16 WMMA GEMM (TDM double-buffered) ----------------
// C[M][N] = A[M][K] * B^T (Bm stored [N][K]) + bias[n]
// mode 0: store f32 to Cout.  mode 1: scatter bf16 into q/k/v [B,H,S,HD].
__global__ __launch_bounds__(256) void mha_gemm_bf16(
    const __bf16* __restrict__ A, const __bf16* __restrict__ Bm,
    const float* __restrict__ bias, float* __restrict__ Cout,
    __bf16* __restrict__ qb, __bf16* __restrict__ kb, __bf16* __restrict__ vb,
    int M, int N, int K, int mode)
{
  __shared__ __align__(16) __bf16 As[2][128 * 32];
  __shared__ __align__(16) __bf16 Bs[2][128 * 32];

  const int tid  = threadIdx.x;
  const int lane = tid & 31;
  const int wave = tid >> 5;
  const int wm = (wave >> 2) * 64;  // 2 waves along M
  const int wn = (wave & 3) * 32;   // 4 waves along N
  const long bm = (long)blockIdx.x * 128;
  const long bn = (long)blockIdx.y * 128;
  const int nk = K >> 5;

  v8f acc[4][2] = {};

#if HAVE_TDM
  const int wv = __builtin_amdgcn_readfirstlane((int)threadIdx.x) >> 5;
  if (wv == 0) {
    tdm_load_2d_bf16(lds_off(&As[0][0]), &A[bm * (long)K], 32, 128,
                     (unsigned long long)K, (unsigned long long)M,
                     (unsigned long long)K, 0, 0, false);
    tdm_load_2d_bf16(lds_off(&Bs[0][0]), &Bm[bn * (long)K], 32, 128,
                     (unsigned long long)K, (unsigned long long)N,
                     (unsigned long long)K, 0, 0, false);
  }
#endif

  for (int ki = 0; ki < nk; ++ki) {
    const int cur = ki & 1;
#if HAVE_TDM
    if (wv == 0) {
      if (ki + 1 < nk) {
        const long k1 = (long)(ki + 1) * 32;
        tdm_load_2d_bf16(lds_off(&As[cur ^ 1][0]), &A[bm * (long)K + k1], 32, 128,
                         (unsigned long long)K, (unsigned long long)M,
                         (unsigned long long)K, 0, 0, false);
        tdm_load_2d_bf16(lds_off(&Bs[cur ^ 1][0]), &Bm[bn * (long)K + k1], 32, 128,
                         (unsigned long long)K, (unsigned long long)N,
                         (unsigned long long)K, 0, 0, false);
        __builtin_amdgcn_s_wait_tensorcnt((short)2);  // two oldest (tile ki) done
      } else {
        __builtin_amdgcn_s_wait_tensorcnt((short)0);
      }
    }
#else
    {
      const int k0 = ki * 32;
      for (int i = 0; i < 2; ++i) {
        int u = tid + i * 256;
        int r = u >> 2;
        int c = (u & 3) * 8;
        *reinterpret_cast<uint4*>(&As[cur][r * 32 + c]) =
            *reinterpret_cast<const uint4*>(&A[(bm + r) * (long)K + k0 + c]);
        *reinterpret_cast<uint4*>(&Bs[cur][r * 32 + c]) =
            *reinterpret_cast<const uint4*>(&Bm[(bn + r) * (long)K + k0 + c]);
      }
    }
#endif
    __syncthreads();

    v16bf af[4], bfr[2];
    for (int i = 0; i < 4; ++i) af[i]  = frag_ld(&As[cur][(wm + i * 16) * 32], 32, lane);
    for (int j = 0; j < 2; ++j) bfr[j] = frag_ld(&Bs[cur][(wn + j * 16) * 32], 32, lane);
    for (int i = 0; i < 4; ++i)
      for (int j = 0; j < 2; ++j)
        acc[i][j] = __builtin_amdgcn_wmma_f32_16x16x32_bf16(
            false, af[i], false, bfr[j], (short)0, acc[i][j], false, false);
    __syncthreads();
  }

  // C fragment layout: element (m,n): vgpr r = m%8, lane = (n%16) + 16*(m/8)
  const int l = lane & 15;
  const int g = (lane >> 4) & 1;
  for (int i = 0; i < 4; ++i)
    for (int j = 0; j < 2; ++j) {
      const long n = bn + wn + j * 16 + l;
      const float bval = bias[n];
      for (int r = 0; r < 8; ++r) {
        const long m = bm + wm + i * 16 + r + 8 * g;
        const float val = acc[i][j][r] + bval;
        if (mode == 0) {
          Cout[m * (long)N + n] = val;
        } else {
          // n -> head h = n/192, t = n%192; which = t/64 (q,k,v); hd = t%64
          const int h     = (int)(n / 192);
          const int t     = (int)(n % 192);
          const int which = t >> 6;
          const int hd    = t & 63;
          const long b = m >> 11;     // m / 2048 (S)
          const long s = m & 2047;    // m % 2048
          __bf16* dst = (which == 0) ? qb : ((which == 1) ? kb : vb);
          dst[(((b * 16 + h) * 2048) + s) * 64 + hd] = (__bf16)val;
        }
      }
    }
}

// ---------------- flash attention (bf16 WMMA, f32 softmax) ----------------
// q,k,v: [B*H][S][HD] bf16. out: attn_bf [B*S][E] bf16 (head-merged).
__global__ __launch_bounds__(128) void mha_attn_bf16(
    const __bf16* __restrict__ q, const __bf16* __restrict__ k,
    const __bf16* __restrict__ v, __bf16* __restrict__ attn_out)
{
  __shared__ __align__(16) __bf16 ks[2][64 * 72];  // K tile [s_k][hd], pitch 72 (TDM pad)
  __shared__ __align__(16) __bf16 vts[64 * 72];    // V^T tile [hd][s_k]
  __shared__ __align__(16) __bf16 ps[64 * 72];     // P tile  [m][s_k]

  const int tid  = threadIdx.x;
  const int wave = tid >> 5;
  const int lane = tid & 31;
  const int l = lane & 15;
  const int g = (lane >> 4) & 1;

  const int qt = blockIdx.x & 31;   // S/64 = 32 q-tiles
  const int bh = blockIdx.x >> 5;   // B*H = 64
  const __bf16* qp  = q + ((long)bh * 2048 + qt * 64) * 64;
  const __bf16* kp0 = k + (long)bh * 2048 * 64;
  const __bf16* vp0 = v + (long)bh * 2048 * 64;
  const int m0 = wave * 16;         // this wave's 16-row strip

  // Q fragments straight from global (row-major [s][hd], 16B-aligned runs)
  v16bf qf[2];
  for (int kk = 0; kk < 2; ++kk) {
    const __bf16* p = qp + (m0 + l) * 64 + kk * 32 + 8 * g;
    uint4* u = reinterpret_cast<uint4*>(&qf[kk]);
    u[0] = *reinterpret_cast<const uint4*>(p);
    u[1] = *reinterpret_cast<const uint4*>(p + 16);
  }

  v8f o[4] = {};
  float rmax[8], rsum[8];
  for (int r = 0; r < 8; ++r) { rmax[r] = -1e30f; rsum[r] = 0.0f; }

#if HAVE_TDM
  const int wv = __builtin_amdgcn_readfirstlane((int)threadIdx.x) >> 5;
  if (wv == 0) {
    // pad_interval code 4 -> every 32 DWORDs (one 64-elem row); pad_amount code 3
    // -> 4 DWORDs (8 elems): LDS row pitch = 72 bf16, matching frag_ld stride.
    tdm_load_2d_bf16(lds_off(&ks[0][0]), kp0, 64, 64, 64ull, 2048ull, 64ull, 4, 3, true);
  }
#endif

#pragma unroll 1
  for (int kt = 0; kt < 32; ++kt) {
    const int cur = kt & 1;
    const __bf16* vp = vp0 + (long)kt * 64 * 64;
    // V tile, stored transposed into LDS (manual: TDM cannot transpose)
    for (int i = 0; i < 4; ++i) {
      int u = tid + i * 128;
      int r = u >> 3;
      int c = (u & 7) * 8;
      uint4 vv = *reinterpret_cast<const uint4*>(&vp[r * 64 + c]);
      const __bf16* ve = reinterpret_cast<const __bf16*>(&vv);
      for (int e = 0; e < 8; ++e) vts[(c + e) * 72 + r] = ve[e];
    }
    __builtin_prefetch(vp + 64 * 64, 0, 1);  // next V tile -> global_prefetch_b8
#if HAVE_TDM
    if (wv == 0) {
      if (kt + 1 < 32) {
        tdm_load_2d_bf16(lds_off(&ks[cur ^ 1][0]), kp0 + (long)(kt + 1) * 64 * 64,
                         64, 64, 64ull, 2048ull, 64ull, 4, 3, true);
        __builtin_amdgcn_s_wait_tensorcnt((short)1);  // tile kt ready
      } else {
        __builtin_amdgcn_s_wait_tensorcnt((short)0);
      }
    }
#else
    {
      const __bf16* kp = kp0 + (long)kt * 64 * 64;
      for (int i = 0; i < 4; ++i) {
        int u = tid + i * 128;
        int r = u >> 3;
        int c = (u & 7) * 8;
        *reinterpret_cast<uint4*>(&ks[cur][r * 72 + c]) =
            *reinterpret_cast<const uint4*>(&kp[r * 64 + c]);
      }
    }
#endif
    __syncthreads();

    // logits strip: 16 x 64, A = Q frag, B = K tile rows (same loader: [N=s_k][K=hd])
    v8f sacc[4];
    for (int j = 0; j < 4; ++j) {
      v8f a = {};
      for (int kk = 0; kk < 2; ++kk) {
        v16bf bfr = frag_ld(&ks[cur][(j * 16) * 72 + kk * 32], 72, lane);
        a = __builtin_amdgcn_wmma_f32_16x16x32_bf16(
            false, qf[kk], false, bfr, (short)0, a, false, false);
      }
      sacc[j] = a * 0.125f;  // 1/sqrt(64)
    }

    // online softmax per row m = m0 + r + 8g (row spread over 16 lanes x 4 frags)
    for (int r = 0; r < 8; ++r) {
      float mval = sacc[0][r];
      for (int j = 1; j < 4; ++j) mval = fmaxf(mval, sacc[j][r]);
      for (int off = 1; off < 16; off <<= 1)
        mval = fmaxf(mval, __shfl_xor(mval, off, 32));
      const float nm = fmaxf(rmax[r], mval);
      const float sc = __expf(rmax[r] - nm);
      rmax[r] = nm;
      float psum = 0.0f;
      for (int j = 0; j < 4; ++j) {
        float pv = __expf(sacc[j][r] - nm);
        sacc[j][r] = pv;
        psum += pv;
      }
      for (int off = 1; off < 16; off <<= 1)
        psum += __shfl_xor(psum, off, 32);
      rsum[r] = rsum[r] * sc + psum;
      for (int j = 0; j < 4; ++j) o[j][r] *= sc;
      const int mrow = m0 + r + 8 * g;
      for (int j = 0; j < 4; ++j)
        ps[mrow * 72 + j * 16 + l] = (__bf16)sacc[j][r];
    }

    // O += P @ V : A = P strip (wave-private rows, same-wave LDS dep), B = V^T rows
    for (int j2 = 0; j2 < 4; ++j2) {
      for (int kk = 0; kk < 2; ++kk) {
        v16bf pa = frag_ld(&ps[m0 * 72 + kk * 32], 72, lane);
        v16bf vb = frag_ld(&vts[(j2 * 16) * 72 + kk * 32], 72, lane);
        o[j2] = __builtin_amdgcn_wmma_f32_16x16x32_bf16(
            false, pa, false, vb, (short)0, o[j2], false, false);
      }
    }
    __syncthreads();
  }

  // epilogue: normalize and write bf16 into merged [B*S][E] layout
  const int b = bh >> 4;   // H = 16
  const int h = bh & 15;
  for (int j2 = 0; j2 < 4; ++j2)
    for (int r = 0; r < 8; ++r) {
      const int srow = qt * 64 + m0 + r + 8 * g;
      const float val = o[j2][r] / rsum[r];
      attn_out[((long)b * 2048 + srow) * 1024 + h * 64 + j2 * 16 + l] = (__bf16)val;
    }
}

// ---------------- host launch ----------------
extern "C" void kernel_launch(void* const* d_in, const int* in_sizes, int n_in,
                              void* d_out, int out_size, void* d_ws, size_t ws_size,
                              hipStream_t stream) {
  (void)in_sizes; (void)n_in; (void)out_size; (void)ws_size;
  const float* x    = (const float*)d_in[0];
  const float* Wqkv = (const float*)d_in[1];
  const float* bqkv = (const float*)d_in[2];
  const float* Wo   = (const float*)d_in[3];
  const float* bo   = (const float*)d_in[4];
  float* out = (float*)d_out;

  char* ws = (char*)d_ws;
  size_t off = 0;
  auto take = [&](size_t bytes) -> void* {
    void* p = ws + off;
    off += (bytes + 255) & ~(size_t)255;
    return p;
  };
  __bf16* x_bf   = (__bf16*)take((size_t)8192 * 1024 * 2);    // x, bf16
  __bf16* wqkvT  = (__bf16*)take((size_t)3072 * 1024 * 2);    // Wqkv^T [3E][D]
  __bf16* woT    = (__bf16*)take((size_t)1024 * 1024 * 2);    // Wo^T   [E][E]
  __bf16* qb     = (__bf16*)take((size_t)64 * 2048 * 64 * 2); // [B*H][S][HD]
  __bf16* kb     = (__bf16*)take((size_t)64 * 2048 * 64 * 2);
  __bf16* vb     = (__bf16*)take((size_t)64 * 2048 * 64 * 2);
  __bf16* att_bf = (__bf16*)take((size_t)8192 * 1024 * 2);    // [B*S][E]

  mha_cvt_bf16<<<2048, 256, 0, stream>>>(x, x_bf, (long)8192 * 1024);
  mha_transpose_bf16<<<2048, 256, 0, stream>>>(Wqkv, wqkvT, 1024, 3072);
  mha_transpose_bf16<<<1024, 256, 0, stream>>>(Wo, woT, 1024, 1024);

  dim3 g1(8192 / 128, 3072 / 128);
  mha_gemm_bf16<<<g1, 256, 0, stream>>>(x_bf, wqkvT, bqkv, nullptr,
                                        qb, kb, vb, 8192, 3072, 1024, 1);

  mha_attn_bf16<<<64 * 32, 128, 0, stream>>>(qb, kb, vb, att_bf);

  dim3 g2(8192 / 128, 1024 / 128);
  mha_gemm_bf16<<<g2, 256, 0, stream>>>(att_bf, woT, bo, out,
                                        nullptr, nullptr, nullptr, 8192, 1024, 1024, 0);
}
